// CausalSelfAttention_9672266351163
// MI455X (gfx1250) — matrix operand order repaired and checked by reading, not compiled, not run on hardware
//
#include <hip/hip_runtime.h>

// ---------------------------------------------------------------------------
// CausalSelfAttention forward for MI455X (gfx1250).
// All matmuls via v_wmma_f32_16x16x32_bf16 (wave32 WMMA).
// Global->LDS staging via Tensor Data Mover (tensor_load_to_lds), double
// buffered with s_wait_tensorcnt; batched ds_load_b128 -> back-to-back WMMAs.
// B=2, T=2048, C=1024, H=16, D=64
// ---------------------------------------------------------------------------

typedef __attribute__((ext_vector_type(16))) __bf16        v16bf;
typedef __attribute__((ext_vector_type(8)))  float         v8f;
typedef __attribute__((ext_vector_type(4)))  unsigned int  v4u;
typedef __attribute__((ext_vector_type(8)))  int           v8i;
typedef __attribute__((ext_vector_type(4)))  int           v4i;

#define T_SEQ 2048
#define N_HEAD 16
#define HEAD_D 64
#define C_EMB 1024

static __device__ __forceinline__ unsigned short f2bf(float f) {
  unsigned int u = __float_as_uint(f);
  u += 0x7FFFu + ((u >> 16) & 1u);           // round-to-nearest-even
  return (unsigned short)(u >> 16);
}
static __device__ __forceinline__ float bf2f(unsigned short h) {
  return __uint_as_float(((unsigned int)h) << 16);
}

union FragU { v16bf v; v4u q[2]; };
// Build a 16-element bf16 fragment from two 16-byte (8 x bf16) chunks.
static __device__ __forceinline__ v16bf load_frag(const unsigned short* lo,
                                                  const unsigned short* hi) {
  FragU f;
  f.q[0] = *(const v4u*)lo;
  f.q[1] = *(const v4u*)hi;
  return f.v;
}

static __device__ __forceinline__ v8f wmma_bf16(v16bf a, v16bf b, v8f c) {
  return __builtin_amdgcn_wmma_f32_16x16x32_bf16(false, a, false, b,
                                                 (short)0, c, false, false);
}

// ---------------------------------------------------------------------------
// Tensor Data Mover: 2D tile (rows x kElems of bf16) global -> LDS.
// D# layout per CDNA5 ISA ch.8 (group0 128b, group1 256b; groups 2/3 zero).
// ---------------------------------------------------------------------------
static __device__ __forceinline__ void tdm_load_2d(unsigned ldsByteOff,
                                                   const void* gptr,
                                                   unsigned kElems,
                                                   unsigned rows,
                                                   unsigned rowStrideElems) {
  unsigned long long ga = (unsigned long long)(__SIZE_TYPE__)gptr;
  v4u g0;
  g0[0] = 1u;                                          // count=1 (valid)
  g0[1] = ldsByteOff;                                  // lds_addr (bytes)
  g0[2] = (unsigned)(ga & 0xFFFFFFFFu);                // global_addr[31:0]
  g0[3] = (unsigned)((ga >> 32) & 0x1FFFFFFu)          // global_addr[56:32]
          | (2u << 30);                                // type=2 ("image")
  v8i g1;
  g1[0] = (int)(1u << 16);                             // data_size=1 (2B), mask=0
  g1[1] = (int)((kElems & 0xFFFFu) << 16);             // tensor_dim0[15:0]
  g1[2] = (int)((kElems >> 16) | ((rows & 0xFFFFu) << 16));   // dim0 hi | dim1 lo
  g1[3] = (int)((rows >> 16) | ((kElems & 0xFFFFu) << 16));   // dim1 hi | tile_dim0
  g1[4] = (int)(rows & 0xFFFFu);                       // tile_dim1 | tile_dim2=0
  g1[5] = (int)rowStrideElems;                         // tensor_dim0_stride lo32
  g1[6] = 0;                                           // stride hi | dim1_stride lo
  g1[7] = 0;
  v4i gz = (v4i){0, 0, 0, 0};
#if defined(__clang_major__) && (__clang_major__ >= 23)
  v8i gz8 = (v8i){0, 0, 0, 0, 0, 0, 0, 0};
  __builtin_amdgcn_tensor_load_to_lds(g0, g1, gz, gz, gz8, 0);
#else
  __builtin_amdgcn_tensor_load_to_lds(g0, g1, gz, gz, 0);
#endif
}

// ---------------------------------------------------------------------------
// Kernel 1a: fp32 -> bf16 elementwise
// ---------------------------------------------------------------------------
__global__ __launch_bounds__(256) void cvt_f32_bf16(const float* __restrict__ src,
                                                    unsigned short* __restrict__ dst,
                                                    int n) {
  int i = blockIdx.x * blockDim.x + threadIdx.x;
  if (i < n) dst[i] = f2bf(src[i]);
}

// ---------------------------------------------------------------------------
// Kernel 1b: W [K,N] fp32 row-major -> Wt [N,K] bf16 (B-transposed for WMMA)
// ---------------------------------------------------------------------------
__global__ __launch_bounds__(256) void transpose_f32_bf16(const float* __restrict__ W,
                                                          unsigned short* __restrict__ Wt,
                                                          int Kdim, int Ndim) {
  int i = blockIdx.x * blockDim.x + threadIdx.x;   // over Ndim*Kdim
  if (i < Ndim * Kdim) {
    int nn = i / Kdim;
    int k  = i - nn * Kdim;
    Wt[i] = f2bf(W[(size_t)k * Ndim + nn]);
  }
}

// ---------------------------------------------------------------------------
// Kernel 2/5: bf16 GEMM  C[M,N] = A[M,K] @ Bt[N,K]^T + bias
// Block = 256 thr (8 waves) -> C tile 128x128.  Per K-step (32):
//   TDM loads A-tile 128x32 and B-tile 128x32 into LDS (double buffered,
//   issued by wave 0, synced with s_wait_tensorcnt + barrier); each wave
//   batches 18 ds_load_b128 then issues 8 back-to-back WMMAs.
// LDS: stage s at byte offset s*16384: [A 8192 B][B 8192 B]; total 32768 B.
// ---------------------------------------------------------------------------
__global__ __launch_bounds__(256) void gemm_bf16_wmma(
    const unsigned short* __restrict__ A,
    const unsigned short* __restrict__ Bt,
    const float* __restrict__ bias,
    float* __restrict__ Cf,            // fp32 output (or null)
    unsigned short* __restrict__ Cb16, // bf16 output (or null)
    int N, int K) {
  extern __shared__ unsigned short smem[];   // dynamic LDS, base offset 0

  const int lane  = threadIdx.x & 31;
  const int wave  = threadIdx.x >> 5;
  const int half  = lane >> 4;
  const int ln    = lane & 15;
  const int tileM = blockIdx.y * 128;
  const int tileN = blockIdx.x * 128;

  const unsigned short* Abase = A + (size_t)tileM * K;
  const unsigned short* Bbase = Bt + (size_t)tileN * K;

  v8f c[8] = {};
  const int nSteps = K >> 5;

  if (wave == 0) {
    tdm_load_2d(0,    Abase, 32, 128, (unsigned)K);
    tdm_load_2d(8192, Bbase, 32, 128, (unsigned)K);
  }

  for (int s = 0; s < nSteps; ++s) {
    const int cur = s & 1;
    if (wave == 0) {
      if (s + 1 < nSteps) {
        const unsigned nextOff = (unsigned)(((s + 1) & 1) * 16384);
        tdm_load_2d(nextOff,        Abase + (s + 1) * 32, 32, 128, (unsigned)K);
        tdm_load_2d(nextOff + 8192, Bbase + (s + 1) * 32, 32, 128, (unsigned)K);
        __builtin_amdgcn_s_wait_tensorcnt((short)2);   // stage s done
      } else {
        __builtin_amdgcn_s_wait_tensorcnt((short)0);
      }
    }
    __syncthreads();   // publish stage `cur`

    const unsigned short* As = smem + cur * 8192;          // 128 x 32
    const unsigned short* Bs = smem + cur * 8192 + 4096;   // 128 x 32

    // Batch ALL fragment loads, then issue WMMAs back-to-back.
    const unsigned short* arow = As + (wave * 16 + ln) * 32 + (half ? 16 : 0);
    v16bf a = load_frag(arow, arow + 8);
    v16bf bfr[8];
#pragma unroll
    for (int n = 0; n < 8; ++n) {
      const unsigned short* bp = Bs + (n * 16 + ln) * 32 + (half ? 16 : 0);
      bfr[n] = load_frag(bp, bp + 8);
    }
#pragma unroll
    for (int n = 0; n < 8; ++n) c[n] = wmma_bf16(a, bfr[n], c[n]);

    __syncthreads();   // all waves done reading before stage is overwritten
  }

  // C layout: lane<16 -> N=lane, VGPR r -> M=r ; lane>=16 -> N=lane-16, M=r+8
#pragma unroll
  for (int n = 0; n < 8; ++n) {
#pragma unroll
    for (int r = 0; r < 8; ++r) {
      int row = tileM + wave * 16 + r + half * 8;
      int col = tileN + n * 16 + ln;
      float v = c[n][r] + bias[col];
      if (Cb16) Cb16[(size_t)row * N + col] = f2bf(v);
      else      Cf[(size_t)row * N + col]   = v;
    }
  }
}

// ---------------------------------------------------------------------------
// Kernel 3: RoPE + split qkv[B,T,3C] (bf16) into
//   q,k : [B,H,T,D] bf16   vT : [B,H,D,T] bf16
// ---------------------------------------------------------------------------
__global__ __launch_bounds__(256) void rope_split(
    const unsigned short* __restrict__ qkvb,
    unsigned short* __restrict__ qb,
    unsigned short* __restrict__ kb,
    unsigned short* __restrict__ vtb) {
  int i = blockIdx.x * blockDim.x + threadIdx.x;  // 0 .. B*H*T*D-1 (4M)
  int d  = i & 63;
  int t  = (i >> 6) & (T_SEQ - 1);
  int bh = i >> 17;
  int b  = bh >> 4;

  size_t rowOff = ((size_t)(b * T_SEQ + t)) * (3 * C_EMB) + (bh & 15) * HEAD_D;
  float q = bf2f(qkvb[rowOff + d]);
  float k = bf2f(qkvb[rowOff + C_EMB + d]);
  float v = bf2f(qkvb[rowOff + 2 * C_EMB + d]);

  int   j   = d & 31;
  float inv = __powf(10000.0f, -(float)(2 * j) / 64.0f);
  float ang = (float)t * inv;
  float cs  = __cosf(ang), sn = __sinf(ang);
  float qr, kr;
  if (d < 32) {
    float q2 = bf2f(qkvb[rowOff + d + 32]);
    float k2 = bf2f(qkvb[rowOff + C_EMB + d + 32]);
    qr = q * cs - q2 * sn;
    kr = k * cs - k2 * sn;
  } else {
    float q1 = bf2f(qkvb[rowOff + d - 32]);
    float k1 = bf2f(qkvb[rowOff + C_EMB + d - 32]);
    qr = q * cs + q1 * sn;
    kr = k * cs + k1 * sn;
  }
  size_t qkIdx = ((size_t)bh * T_SEQ + t) * HEAD_D + d;
  qb[qkIdx] = f2bf(qr);
  kb[qkIdx] = f2bf(kr);
  vtb[((size_t)bh * HEAD_D + d) * T_SEQ + t] = f2bf(v);
}

// ---------------------------------------------------------------------------
// Kernel 4: flash attention. One wave per (b,h, 16-query tile).
// K-tile (32x64) and V^T-tile (64x32) prefetched to LDS by TDM, double
// buffered with s_wait_tensorcnt (single wave => no barriers). 32 keys per
// step: batched fragment loads, 4 score WMMAs + 4 P@V WMMAs, online softmax
// via shfl_xor, P relayout through LDS.
// LDS (dynamic): stage s at s*8192: [K 4096][V 4096]; pbuf at 16384 (1 KB).
// ---------------------------------------------------------------------------
__global__ __launch_bounds__(32) void flash_attn(
    const unsigned short* __restrict__ qb,
    const unsigned short* __restrict__ kb,
    const unsigned short* __restrict__ vtb,
    unsigned short* __restrict__ ybf) {
  extern __shared__ unsigned short smem[];
  unsigned short* pbuf = smem + 8192;         // byte offset 16384

  const int lane  = threadIdx.x;
  const int half  = lane >> 4;
  const int ln    = lane & 15;
  const int qBase = blockIdx.x * 16;
  const int bh    = blockIdx.y;
  const int b     = bh >> 4;
  const int h     = bh & 15;
  const float scale = 0.125f;   // 1/sqrt(64)

  const unsigned short* kBase = kb  + (size_t)bh * T_SEQ * HEAD_D;
  const unsigned short* vBase = vtb + (size_t)bh * HEAD_D * T_SEQ;

  // Q fragments: rows qBase..qBase+15, d = 0..31 and 32..63
  const unsigned short* qrow =
      qb + ((size_t)bh * T_SEQ + qBase + ln) * HEAD_D + (half ? 16 : 0);
  v16bf aQ0 = load_frag(qrow, qrow + 8);
  v16bf aQ1 = load_frag(qrow + 32, qrow + 40);

  v8f o[4] = {};
  float m[8], l[8];
#pragma unroll
  for (int r = 0; r < 8; ++r) { m[r] = -1e30f; l[r] = 0.0f; }

  const int nChunks = (qBase + 16 + 31) >> 5;

  // prefetch chunk 0:  K rows kv0..kv0+31 (contiguous), V rows d=0..63 stride T
  tdm_load_2d(0,    kBase, 64, 32, 64);
  tdm_load_2d(4096, vBase, 32, 64, T_SEQ);

  for (int j = 0; j < nChunks; ++j) {
    const int cur = j & 1;
    if (j + 1 < nChunks) {
      const int kvn = (j + 1) * 32;
      const unsigned nextOff = (unsigned)(((j + 1) & 1) * 8192);
      tdm_load_2d(nextOff,        kBase + (size_t)kvn * HEAD_D, 64, 32, 64);
      tdm_load_2d(nextOff + 4096, vBase + kvn, 32, 64, T_SEQ);
      __builtin_amdgcn_s_wait_tensorcnt((short)2);
    } else {
      __builtin_amdgcn_s_wait_tensorcnt((short)0);
    }

    const int kv0 = j * 32;
    const unsigned short* Ks = smem + cur * 4096;          // [32 kv][64 d]
    const unsigned short* Vs = smem + cur * 4096 + 2048;   // [64 d][32 kv]

    // Batch the 4 K fragments, then 4 back-to-back score WMMAs.
    v16bf kf[4];
    {
      const unsigned short* kp0 = Ks + ln * 64 + (half ? 16 : 0);
      const unsigned short* kp1 = Ks + (16 + ln) * 64 + (half ? 16 : 0);
      kf[0] = load_frag(kp0, kp0 + 8);
      kf[1] = load_frag(kp0 + 32, kp0 + 40);
      kf[2] = load_frag(kp1, kp1 + 8);
      kf[3] = load_frag(kp1 + 32, kp1 + 40);
    }
    v8f s0 = {}, s1 = {};
    s0 = wmma_bf16(aQ0, kf[0], s0);
    s0 = wmma_bf16(aQ1, kf[1], s0);
    s1 = wmma_bf16(aQ0, kf[2], s1);
    s1 = wmma_bf16(aQ1, kf[3], s1);

    float p0[8], p1[8];
#pragma unroll
    for (int r = 0; r < 8; ++r) {
      const int row = qBase + r + half * 8;
      float v0 = s0[r] * scale;
      float v1 = s1[r] * scale;
      if (kv0 + ln > row)      v0 = -1e30f;   // causal mask
      if (kv0 + 16 + ln > row) v1 = -1e30f;
      float t = fmaxf(v0, v1);
      t = fmaxf(t, __shfl_xor(t, 1, 32));
      t = fmaxf(t, __shfl_xor(t, 2, 32));
      t = fmaxf(t, __shfl_xor(t, 4, 32));
      t = fmaxf(t, __shfl_xor(t, 8, 32));
      const float mn = fmaxf(m[r], t);
      p0[r] = __expf(v0 - mn);
      p1[r] = __expf(v1 - mn);
      float rs = p0[r] + p1[r];
      rs += __shfl_xor(rs, 1, 32);
      rs += __shfl_xor(rs, 2, 32);
      rs += __shfl_xor(rs, 4, 32);
      rs += __shfl_xor(rs, 8, 32);
      const float alpha = __expf(m[r] - mn);
      l[r] = l[r] * alpha + rs;
      m[r] = mn;
#pragma unroll
      for (int n = 0; n < 4; ++n) o[n][r] = o[n][r] * alpha;
    }

    // P (C-layout f32) -> LDS [16 q][32 kv] bf16
#pragma unroll
    for (int r = 0; r < 8; ++r) {
      pbuf[(r + half * 8) * 32 + ln]      = f2bf(p0[r]);
      pbuf[(r + half * 8) * 32 + 16 + ln] = f2bf(p1[r]);
    }
    // single wave: LDS ops are in-order; batch aP + 4 V fragments, then WMMAs
    const unsigned short* pl = pbuf + ln * 32 + (half ? 16 : 0);
    v16bf aP = load_frag(pl, pl + 8);
    v16bf vf[4];
#pragma unroll
    for (int n = 0; n < 4; ++n) {
      const unsigned short* vp = Vs + (n * 16 + ln) * 32 + (half ? 16 : 0);
      vf[n] = load_frag(vp, vp + 8);
    }
#pragma unroll
    for (int n = 0; n < 4; ++n) o[n] = wmma_bf16(aP, vf[n], o[n]);
  }

  // y[B,T,C] bf16, c = h*64 + d
#pragma unroll
  for (int n = 0; n < 4; ++n) {
#pragma unroll
    for (int r = 0; r < 8; ++r) {
      const int row = qBase + r + half * 8;
      const int col = h * HEAD_D + n * 16 + ln;
      ybf[((size_t)b * T_SEQ + row) * C_EMB + col] = f2bf(o[n][r] / l[r]);
    }
  }
}

// ---------------------------------------------------------------------------
// Host-side orchestration
// ---------------------------------------------------------------------------
extern "C" void kernel_launch(void* const* d_in, const int* in_sizes, int n_in,
                              void* d_out, int out_size, void* d_ws, size_t ws_size,
                              hipStream_t stream) {
  (void)in_sizes; (void)n_in; (void)out_size; (void)ws_size;
  const float* x     = (const float*)d_in[0];   // [2,2048,1024]
  const float* Wqkv  = (const float*)d_in[1];   // [1024,3072]
  const float* bqkv  = (const float*)d_in[2];   // [3072]
  const float* Wproj = (const float*)d_in[3];   // [1024,1024]
  const float* bproj = (const float*)d_in[4];   // [1024]
  float* out = (float*)d_out;

  char* ws = (char*)d_ws;
  unsigned short* xb     = (unsigned short*)(ws);              //  8 MiB
  unsigned short* wqkvT  = (unsigned short*)(ws +  8388608);   //  6 MiB [3072,1024]
  unsigned short* wprojT = (unsigned short*)(ws + 14680064);   //  2 MiB [1024,1024]
  unsigned short* qkvb   = (unsigned short*)(ws + 16777216);   // 24 MiB [4096,3072]
  unsigned short* qbuf   = (unsigned short*)(ws + 41943040);   //  8 MiB [B,H,T,D]
  unsigned short* kbuf   = (unsigned short*)(ws + 50331648);   //  8 MiB [B,H,T,D]
  unsigned short* vtbuf  = (unsigned short*)(ws + 58720256);   //  8 MiB [B,H,D,T]
  unsigned short* ybf    = (unsigned short*)(ws + 67108864);   //  8 MiB [4096,1024]

  const int M = 2 * T_SEQ;              // 4096 rows

  // 1) precision / layout conversion
  cvt_f32_bf16<<<(M * C_EMB) / 256, 256, 0, stream>>>(x, xb, M * C_EMB);
  transpose_f32_bf16<<<(3 * C_EMB * C_EMB) / 256, 256, 0, stream>>>(Wqkv, wqkvT, C_EMB, 3 * C_EMB);
  transpose_f32_bf16<<<(C_EMB * C_EMB) / 256, 256, 0, stream>>>(Wproj, wprojT, C_EMB, C_EMB);

  // 2) qkv = x @ Wqkv + bqkv  (bf16 out), TDM-staged WMMA GEMM (128x128 tiles)
  gemm_bf16_wmma<<<dim3((3 * C_EMB) / 128, M / 128), 256, 32768, stream>>>(
      xb, wqkvT, bqkv, nullptr, qkvb, 3 * C_EMB, C_EMB);

  // 3) RoPE + head split / V transpose
  rope_split<<<(2 * N_HEAD * T_SEQ * HEAD_D) / 256, 256, 0, stream>>>(qkvb, qbuf, kbuf, vtbuf);

  // 4) causal flash attention -> y bf16 [B,T,C]
  flash_attn<<<dim3(T_SEQ / 16, 2 * N_HEAD), 32, 17408, stream>>>(qbuf, kbuf, vtbuf, ybf);

  // 5) out = y @ Wproj + bproj  (fp32 out), TDM-staged WMMA GEMM
  gemm_bf16_wmma<<<dim3(C_EMB / 128, M / 128), 256, 32768, stream>>>(
      ybf, wprojT, bproj, out, nullptr, C_EMB, C_EMB);
}
